// SingleLayerGCNNetwork_41918880809421
// MI455X (gfx1250) — matrix operand order repaired and checked by reading, not compiled, not run on hardware
//
#include <hip/hip_runtime.h>
#include <hip/hip_bf16.h>

// ---------------------------------------------------------------------------
// 3-layer GCN for MI455X (gfx1250, wave32).
//  * GEMMs via V_WMMA_F32_16X16X32_BF16 (bf16 in, f32 accum), register-blocked
//    5 M-tiles per wave so each weight (B) fragment is loaded once per K-step
//    and reused by 5 WMMAs.  Weights pre-transposed so B frags are contiguous.
//  * Edge aggregation via f32 global atomics; h matrices (<=153.6MB) are
//    L2-resident on the 192MB L2, so the scatter runs at L2 bandwidth.
// ---------------------------------------------------------------------------

typedef __attribute__((ext_vector_type(16))) __bf16 bf16x16;
typedef __attribute__((ext_vector_type(8)))  float  f32x8;
typedef __attribute__((ext_vector_type(4)))  float  f32x4;

// ------------------------------ helpers ------------------------------------

__global__ void zero_f32_kernel(float* __restrict__ p, size_t n) {
    size_t i = blockIdx.x * (size_t)blockDim.x + threadIdx.x;
    if (i < n) p[i] = 0.0f;
}

__global__ void fill_f32_kernel(float* __restrict__ p, float v, size_t n) {
    size_t i = blockIdx.x * (size_t)blockDim.x + threadIdx.x;
    if (i < n) p[i] = v;
}

// deg[dst] += 1 for every edge (deg buffer pre-zeroed)
__global__ void degree_kernel(const int* __restrict__ dst, float* __restrict__ deg, int E) {
    int e = blockIdx.x * blockDim.x + threadIdx.x;
    if (e < E) atomicAdd(&deg[dst[e]], 1.0f);
}

// in-place deg -> deg^{-1/2} (self-loop adds 1)
__global__ void dis_kernel(float* __restrict__ deg, int n) {
    int i = blockIdx.x * blockDim.x + threadIdx.x;
    if (i < n) deg[i] = rsqrtf(1.0f + deg[i]);
}

// gather subgraph rows of building_x and convert f32 -> bf16
__global__ void gather_bf16_kernel(const float* __restrict__ x, const int* __restrict__ l2g,
                                   __bf16* __restrict__ out, int n, int F) {
    size_t idx = blockIdx.x * (size_t)blockDim.x + threadIdx.x;
    if (idx >= (size_t)n * F) return;
    int i = (int)(idx / F), f = (int)(idx % F);
    out[idx] = (__bf16)x[(size_t)l2g[i] * F + f];
}

// W [K,N] f32 (row-major) -> Wt [N,K] bf16
__global__ void transpose_bf16_kernel(const float* __restrict__ W, __bf16* __restrict__ Wt,
                                      int K, int N) {
    size_t idx = blockIdx.x * (size_t)blockDim.x + threadIdx.x;
    if (idx >= (size_t)K * N) return;
    int n = (int)(idx / K), k = (int)(idx % K);
    Wt[idx] = (__bf16)W[(size_t)k * N + n];
}

// W3 [K,OUTC] f32 -> Wt3 [16,K] bf16, rows >= OUTC zero-padded
__global__ void pad_w3_kernel(const float* __restrict__ W, __bf16* __restrict__ Wt,
                              int K, int OUTC) {
    size_t idx = blockIdx.x * (size_t)blockDim.x + threadIdx.x;
    if (idx >= (size_t)16 * K) return;
    int n = (int)(idx / K), k = (int)(idx % K);
    Wt[idx] = (__bf16)((n < OUTC) ? W[(size_t)k * OUTC + n] : 0.0f);
}

// ------------------------------ WMMA GEMM ----------------------------------
// C[M,N] (f32) = A[M,K] (bf16, row-major) x Bt[N,K]^T (bf16, row-major = B^T)
// One wave computes MBLK vertically-stacked 16x16 tiles, sharing B fragments.
// M,N multiples of 16; K multiple of 32.  All branches are wave-uniform so
// EXEC is all-ones around every WMMA (ISA requirement).
template <int MBLK>
__global__ void gemm_bf16_wmma(const __bf16* __restrict__ A,
                               const __bf16* __restrict__ Bt,
                               float* __restrict__ C,
                               int M, int N, int K) {
    const int wavesPerBlock = blockDim.x >> 5;
    const int wave    = blockIdx.x * wavesPerBlock + (threadIdx.x >> 5);
    const int nTiles  = N >> 4;
    const int mTiles  = M >> 4;
    const int mGroups = (mTiles + MBLK - 1) / MBLK;
    if (wave >= mGroups * nTiles) return;          // wave-uniform
    const int mg     = wave / nTiles;
    const int nTile  = wave % nTiles;
    const int mTile0 = mg * MBLK;
    const int lane   = threadIdx.x & 31;
    const int l15    = lane & 15;
    const int khalf  = lane >> 4;

    const __bf16* bRow = Bt + (size_t)(nTile * 16 + l15) * K;

    const __bf16* aRow[MBLK];
    bool valid[MBLK];
#pragma unroll
    for (int i = 0; i < MBLK; ++i) {
        valid[i] = (mTile0 + i) < mTiles;
        const int mt = valid[i] ? (mTile0 + i) : mTile0;   // clamp: stay in bounds
        aRow[i] = A + (size_t)(mt * 16 + l15) * K;
    }

    f32x8 acc[MBLK] = {};
    for (int kb = 0; kb < K; kb += 32) {
        // B 32x16 bf16 fragment: lane holds col n=l15, K = kb+16h .. kb+16h+15
        union { bf16x16 v; f32x4 q[2]; } fb;
        fb.q[0] = *(const f32x4*)(bRow + kb + khalf * 16);
        fb.q[1] = *(const f32x4*)(bRow + kb + khalf * 16 + 8);
#pragma unroll
        for (int i = 0; i < MBLK; ++i) {
            // A 16x32 bf16 fragment: lane holds row m; dwords 0-3 = K kb+16h..+7,
            // dwords 4-7 = K kb+16h+8..+15  (h = khalf)
            union { bf16x16 v; f32x4 q[2]; } fa;
            fa.q[0] = *(const f32x4*)(aRow[i] + kb + khalf * 16);
            fa.q[1] = *(const f32x4*)(aRow[i] + kb + khalf * 16 + 8);
            acc[i] = __builtin_amdgcn_wmma_f32_16x16x32_bf16(false, fa.v, false, fb.v,
                                                             (short)0, acc[i], false, false);
        }
    }
    // D layout: VGPR r -> M = tile*16 + khalf*8 + r, N = nTile*16 + l15
#pragma unroll
    for (int i = 0; i < MBLK; ++i) {
        if (!valid[i]) continue;                   // wave-uniform
        float* cp = C + (size_t)((mTile0 + i) * 16 + khalf * 8) * N + (nTile * 16 + l15);
#pragma unroll
        for (int r = 0; r < 8; ++r) cp[(size_t)r * N] = acc[i][r];
    }
}

// --------------------------- edge aggregation ------------------------------
// agg[dst, :] += dis[src]*dis[dst] * h[src, :]   (f32 atomics, L2-resident)
template <int VEC>
__global__ void edge_scatter_kernel(const float* __restrict__ h, int hs,
                                    const int* __restrict__ src, const int* __restrict__ dst,
                                    const float* __restrict__ dis,
                                    float* __restrict__ agg, int OUTF, int E) {
    size_t idx = blockIdx.x * (size_t)blockDim.x + threadIdx.x;
    const int chunks = OUTF / VEC;
    if (idx >= (size_t)E * chunks) return;
    int e = (int)(idx / chunks);
    int c = (int)(idx % chunks) * VEC;
    int s = src[e], d = dst[e];
    float w = dis[s] * dis[d];
    const float* hp = h + (size_t)s * hs + c;
    float* ap = agg + (size_t)d * OUTF + c;
#pragma unroll
    for (int i = 0; i < VEC; ++i) atomicAdd(ap + i, hp[i] * w);
}

// act = relu(agg + h*dis^2 + b), stored bf16 for the next WMMA GEMM
__global__ void bias_act_kernel(const float* __restrict__ agg, const float* __restrict__ h,
                                const float* __restrict__ dis, const float* __restrict__ b,
                                __bf16* __restrict__ act, int n, int OUTF) {
    size_t idx = blockIdx.x * (size_t)blockDim.x + threadIdx.x;
    if (idx >= (size_t)n * OUTF) return;
    int i = (int)(idx / OUTF), f = (int)(idx % OUTF);
    float dv = dis[i];
    float v = agg[idx] + h[idx] * dv * dv + b[f];
    act[idx] = (__bf16)fmaxf(v, 0.0f);
}

// final: logits = agg + h*dis^2 + b3; log_softmax over 2 classes; scatter to out
__global__ void final_kernel(const float* __restrict__ agg, const float* __restrict__ h, int hs,
                             const float* __restrict__ dis, const float* __restrict__ b,
                             const int* __restrict__ l2g, float* __restrict__ out, int n) {
    int i = blockIdx.x * blockDim.x + threadIdx.x;
    if (i >= n) return;
    float d2 = dis[i] * dis[i];
    float v0 = agg[(size_t)i * 2 + 0] + h[(size_t)i * hs + 0] * d2 + b[0];
    float v1 = agg[(size_t)i * 2 + 1] + h[(size_t)i * hs + 1] * d2 + b[1];
    float m = fmaxf(v0, v1);
    float lse = m + logf(expf(v0 - m) + expf(v1 - m));
    int g = l2g[i];
    out[(size_t)g * 2 + 0] = v0 - lse;
    out[(size_t)g * 2 + 1] = v1 - lse;
}

// ------------------------------- launch ------------------------------------

extern "C" void kernel_launch(void* const* d_in, const int* in_sizes, int n_in,
                              void* d_out, int out_size, void* d_ws, size_t ws_size,
                              hipStream_t stream) {
    const float* x   = (const float*)d_in[0];
    const int*   l2g = (const int*)d_in[1];
    const int*   esr = (const int*)d_in[2];
    const int*   eds = (const int*)d_in[3];
    const float* W1  = (const float*)d_in[4];
    const float* b1  = (const float*)d_in[5];
    const float* W2  = (const float*)d_in[6];
    const float* b2  = (const float*)d_in[7];
    const float* W3  = (const float*)d_in[8];
    const float* b3  = (const float*)d_in[9];
    float* out = (float*)d_out;

    const int NSUB = in_sizes[1];                 // 150000 (multiple of 16)
    const int E    = in_sizes[2];                 // 800000
    const int H1   = in_sizes[5];                 // 256
    const int H2   = in_sizes[7];                 // 128
    const int OUTC = in_sizes[9];                 // 2
    const int FIN  = in_sizes[4] / H1;            // 128
    const int NTOT = in_sizes[0] / FIN;           // 200000

    constexpr int MBLK = 5;                       // 9375 m-tiles = 1875 groups, exact

    // workspace layout
    char* wsp = (char*)d_ws;
    size_t off = 0;
    auto take = [&](size_t bytes) -> void* {
        void* p = wsp + off;
        off += (bytes + 255) & ~(size_t)255;
        return p;
    };
    float*  dis = (float*)take((size_t)NSUB * 4);              // deg -> deg^{-1/2}
    __bf16* X   = (__bf16*)take((size_t)NSUB * H1 * 2);        // bf16 activations (reused all layers)
    float*  H   = (float*)take((size_t)NSUB * H1 * 4);         // GEMM output h (f32)
    float*  G   = (float*)take((size_t)NSUB * H1 * 4);         // scatter accumulator
    __bf16* Wt1 = (__bf16*)take((size_t)H1 * FIN * 2);
    __bf16* Wt2 = (__bf16*)take((size_t)H2 * H1 * 2);
    __bf16* Wt3 = (__bf16*)take((size_t)16 * H2 * 2);

    const int T = 256;                                          // 8 waves / block
    auto nb = [&](size_t n) -> unsigned { return (unsigned)((n + T - 1) / T); };
    auto gemmBlocks = [&](int M, int N) -> unsigned {
        size_t mGroups = ((size_t)(M / 16) + MBLK - 1) / MBLK;
        size_t waves = mGroups * (size_t)(N / 16);
        return (unsigned)((waves + 7) / 8);
    };

    // ---- normalization: dis = (1 + in_degree)^{-1/2}
    zero_f32_kernel<<<nb(NSUB), T, 0, stream>>>(dis, (size_t)NSUB);
    degree_kernel<<<nb(E), T, 0, stream>>>(eds, dis, E);
    dis_kernel<<<nb(NSUB), T, 0, stream>>>(dis, NSUB);

    // ---- input gather + weight prep (bf16, B pre-transposed)
    gather_bf16_kernel<<<nb((size_t)NSUB * FIN), T, 0, stream>>>(x, l2g, X, NSUB, FIN);
    transpose_bf16_kernel<<<nb((size_t)FIN * H1), T, 0, stream>>>(W1, Wt1, FIN, H1);
    transpose_bf16_kernel<<<nb((size_t)H1 * H2), T, 0, stream>>>(W2, Wt2, H1, H2);
    pad_w3_kernel<<<nb((size_t)16 * H2), T, 0, stream>>>(W3, Wt3, H2, OUTC);

    // ---- layer 1: 150000x128 @ 128x256
    gemm_bf16_wmma<MBLK><<<gemmBlocks(NSUB, H1), T, 0, stream>>>(X, Wt1, H, NSUB, H1, FIN);
    zero_f32_kernel<<<nb((size_t)NSUB * H1), T, 0, stream>>>(G, (size_t)NSUB * H1);
    edge_scatter_kernel<4><<<nb((size_t)E * (H1 / 4)), T, 0, stream>>>(H, H1, esr, eds, dis, G, H1, E);
    bias_act_kernel<<<nb((size_t)NSUB * H1), T, 0, stream>>>(G, H, dis, b1, X, NSUB, H1);

    // ---- layer 2: 150000x256 @ 256x128
    gemm_bf16_wmma<MBLK><<<gemmBlocks(NSUB, H2), T, 0, stream>>>(X, Wt2, H, NSUB, H2, H1);
    zero_f32_kernel<<<nb((size_t)NSUB * H2), T, 0, stream>>>(G, (size_t)NSUB * H2);
    edge_scatter_kernel<4><<<nb((size_t)E * (H2 / 4)), T, 0, stream>>>(H, H2, esr, eds, dis, G, H2, E);
    bias_act_kernel<<<nb((size_t)NSUB * H2), T, 0, stream>>>(G, H, dis, b2, X, NSUB, H2);

    // ---- layer 3: 150000x128 @ 128x2 (N padded to 16 for WMMA; h row stride 16)
    gemm_bf16_wmma<MBLK><<<gemmBlocks(NSUB, 16), T, 0, stream>>>(X, Wt3, H, NSUB, 16, H2);
    zero_f32_kernel<<<nb((size_t)NSUB * OUTC), T, 0, stream>>>(G, (size_t)NSUB * OUTC);
    edge_scatter_kernel<2><<<nb((size_t)E * (OUTC / 2)), T, 0, stream>>>(H, 16, esr, eds, dis, G, OUTC, E);

    // ---- output: fill with log_softmax(0,0) = -ln2, scatter subgraph rows
    fill_f32_kernel<<<nb((size_t)NTOT * OUTC), T, 0, stream>>>(out, -0.69314718055994531f,
                                                              (size_t)NTOT * OUTC);
    final_kernel<<<nb(NSUB), T, 0, stream>>>(G, H, 16, dis, b3, l2g, out, NSUB);
}
